// GatedResnet_60292750901429
// MI455X (gfx1250) — matrix operand order repaired and checked by reading, not compiled, not run on hardware
//
#include <hip/hip_runtime.h>

typedef __attribute__((ext_vector_type(16))) __bf16        v16bf;
typedef __attribute__((ext_vector_type(8)))  float         v8f;
typedef __attribute__((ext_vector_type(4)))  unsigned int  u32x4;

union Frag { struct { u32x4 lo, hi; } q; v16bf v; };

__device__ __forceinline__ unsigned short f2bf(float f) {
  union { float f; unsigned int u; } a; a.f = f;
  unsigned int u = a.u;
  u += 0x7FFFu + ((u >> 16) & 1u);   // round-to-nearest-even
  return (unsigned short)(u >> 16);
}

// ---------------- elu(x) -> bf16 ----------------
__global__ __launch_bounds__(256) void elu_to_bf16(const float* __restrict__ x,
                                                   unsigned short* __restrict__ o, int n) {
  int i = blockIdx.x * blockDim.x + threadIdx.x;
  if (i < n) {
    float v = x[i];
    v = v > 0.f ? v : (__expf(v) - 1.f);
    o[i] = f2bf(v);
  }
}

// ---- pack conv weights [3,3,CIN,COUT] f32 -> WMMA B-fragment layout bf16 ----
// layout idx = ((((tap*NT + nt)*KC + kc)*32 + lane)*8 + v); lane: n = nt*16+(lane&15),
// g = lane>>4; dword v holds bf16 pair for k = kc*32 + (v<4 ? 8*g + 2*v : 16 + 8*g + 2*(v-4)) (+1)
__global__ __launch_bounds__(256) void pack_weights(const float* __restrict__ Wsrc,
                                                    unsigned int* __restrict__ dst,
                                                    int CIN, int COUT) {
  int idx = blockIdx.x * blockDim.x + threadIdx.x;
  int KC = CIN >> 5, NT = COUT >> 4;
  int total = 9 * NT * KC * 32 * 8;
  if (idx >= total) return;
  int v    = idx & 7;
  int lane = (idx >> 3) & 31;
  int t    = idx >> 8;
  int kc   = t % KC; t /= KC;
  int nt   = t % NT; t /= NT;
  int tap  = t;
  int n = nt * 16 + (lane & 15);
  int g = lane >> 4;
  int k = kc * 32 + (v < 4 ? 8 * g + 2 * v : 16 + 8 * g + 2 * (v - 4));
  float w0 = Wsrc[((size_t)tap * CIN + k    ) * COUT + n];
  float w1 = Wsrc[((size_t)tap * CIN + k + 1) * COUT + n];
  dst[idx] = (unsigned int)f2bf(w0) | ((unsigned int)f2bf(w1) << 16);
}

// ---------------- implicit-GEMM 3x3 'same' conv via WMMA bf16 ----------------
// block = 256 thr = 8 waves; each wave owns 16 pixels x 64 couts (4 accumulators);
// block tile = 128 pixels x 64 couts. One A fragment feeds 4 WMMAs.
template <int CIN, int COUT, bool OUT_BF16>
__global__ __launch_bounds__(256) void conv3x3_wmma(
    const unsigned short* __restrict__ in,   // [P, CIN] bf16
    const unsigned int*  __restrict__ wpk,   // packed weights
    const float* __restrict__ bias,
    void* __restrict__ outv,                 // [P, COUT] bf16 or f32
    int H, int W) {
  constexpr int KC = CIN / 32;
  constexpr int NT = COUT / 16;
  const int lane = threadIdx.x & 31;
  const int wave = threadIdx.x >> 5;           // 0..7 -> pixel-row tile
  const int nBlksN = COUT / 64;
  const int mblk = blockIdx.x / nBlksN;
  const int nblk = blockIdx.x - mblk * nBlksN;
  const int p0 = mblk * 128 + wave * 16;       // 16-pixel row base for this wave
  const int nt0 = nblk * 4;                    // 4 consecutive 16-col tiles
  const int ncol0 = nt0 * 16;
  const int m = lane & 15;                     // A row / D col within tile
  const int g = lane >> 4;                     // K-group
  const int HW = H * W;
  const int p = p0 + m;
  const int bb = p / HW;
  const int rem = p - bb * HW;
  const int y = rem / W;
  const int x = rem - y * W;

  const u32x4 zero = {0u, 0u, 0u, 0u};
  v8f acc0 = {}, acc1 = {}, acc2 = {}, acc3 = {};
  for (int tap = 0; tap < 9; ++tap) {
    const int dy = tap / 3 - 1;
    const int dx = tap - (tap / 3) * 3 - 1;
    const int yy = y + dy, xx = x + dx;
    const bool valid = ((unsigned)yy < (unsigned)H) && ((unsigned)xx < (unsigned)W);
    const unsigned short* ab = in + ((size_t)((bb * H + yy) * W + xx) * CIN + 8 * g);
    const unsigned int*   wt = wpk + (((size_t)tap * NT + nt0) * KC * 32 + lane) * 8;
#pragma unroll
    for (int kc = 0; kc < KC; ++kc) {
      Frag a;
      if (valid) {
        a.q.lo = *(const u32x4*)(ab + kc * 32);        // channels 8g..8g+7
        a.q.hi = *(const u32x4*)(ab + kc * 32 + 16);   // channels 16+8g..16+8g+7
      } else { a.q.lo = zero; a.q.hi = zero; }
      Frag b0, b1, b2, b3;
      const unsigned int* wk = wt + kc * 256;
      b0.q.lo = *(const u32x4*)(wk);
      b0.q.hi = *(const u32x4*)(wk + 4);
      b1.q.lo = *(const u32x4*)(wk + KC * 256);
      b1.q.hi = *(const u32x4*)(wk + KC * 256 + 4);
      b2.q.lo = *(const u32x4*)(wk + 2 * KC * 256);
      b2.q.hi = *(const u32x4*)(wk + 2 * KC * 256 + 4);
      b3.q.lo = *(const u32x4*)(wk + 3 * KC * 256);
      b3.q.hi = *(const u32x4*)(wk + 3 * KC * 256 + 4);
      acc0 = __builtin_amdgcn_wmma_f32_16x16x32_bf16(false, a.v, false, b0.v,
                                                     (short)0, acc0, false, false);
      acc1 = __builtin_amdgcn_wmma_f32_16x16x32_bf16(false, a.v, false, b1.v,
                                                     (short)0, acc1, false, false);
      acc2 = __builtin_amdgcn_wmma_f32_16x16x32_bf16(false, a.v, false, b2.v,
                                                     (short)0, acc2, false, false);
      acc3 = __builtin_amdgcn_wmma_f32_16x16x32_bf16(false, a.v, false, b3.v,
                                                     (short)0, acc3, false, false);
    }
  }
  // D layout: lane -> col = m; VGPR r -> row = r + 8*g
  v8f accs[4] = {acc0, acc1, acc2, acc3};
#pragma unroll
  for (int j = 0; j < 4; ++j) {
    const float bv = bias[ncol0 + j * 16 + m];
#pragma unroll
    for (int r = 0; r < 8; ++r) {
      const float val = accs[j][r] + bv;
      const size_t oi = (size_t)(p0 + r + 8 * g) * COUT + ncol0 + j * 16 + m;
      if (OUT_BF16) ((unsigned short*)outv)[oi] = f2bf(val);
      else          ((float*)outv)[oi] = val;
    }
  }
}

// ---------------- c3 = a * sigmoid(b) ----------------
__global__ __launch_bounds__(256) void gate_kernel(const float* __restrict__ c2,
                                                   float* __restrict__ c3, int P) {
  int i = blockIdx.x * blockDim.x + threadIdx.x;
  if (i >= P * 128) return;
  int p = i >> 7, c = i & 127;
  float a = c2[(size_t)p * 256 + c];
  float b = c2[(size_t)p * 256 + 128 + c];
  c3[i] = a * (1.f / (1.f + __expf(-b)));
}

// ---------------- q/k/v projections (K=128, 24 total cols: VALU) ----------------
__global__ __launch_bounds__(256) void qkv_kernel(const float* __restrict__ c3,
    const float* __restrict__ Wq, const float* __restrict__ bq,
    const float* __restrict__ Wk, const float* __restrict__ bk,
    const float* __restrict__ Wv, const float* __restrict__ bv,
    float* __restrict__ q, float* __restrict__ k, float* __restrict__ v, int P) {
  int i = blockIdx.x * blockDim.x + threadIdx.x;
  if (i >= P * 24) return;
  int n = i / 24, col = i - n * 24;
  const float* Wp; float bias; float* dst; int stride, cc;
  if (col < 4)      { Wp = Wq; cc = col;     bias = bq[cc]; dst = q + (size_t)n * 4  + cc; stride = 4;  }
  else if (col < 8) { Wp = Wk; cc = col - 4; bias = bk[cc]; dst = k + (size_t)n * 4  + cc; stride = 4;  }
  else              { Wp = Wv; cc = col - 8; bias = bv[cc]; dst = v + (size_t)n * 16 + cc; stride = 16; }
  const float* f = c3 + (size_t)n * 128;
  float acc = bias;
#pragma unroll 8
  for (int kk = 0; kk < 128; ++kk) acc += f[kk] * Wp[kk * stride + cc];
  *dst = acc;
}

// ---------------- strict-causal flash attention, one wave per query row ----------------
__global__ __launch_bounds__(256) void attn_kernel(const float* __restrict__ q,
                                                   const float* __restrict__ k,
                                                   const float* __restrict__ v,
                                                   float* __restrict__ atten, int Nn) {
  const int lane = threadIdx.x & 31;
  const int wid  = (blockIdx.x * blockDim.x + threadIdx.x) >> 5;
  const int b = wid / Nn;
  const int i = wid - b * Nn;
  const size_t rb = (size_t)b * Nn;

  const float* qp = q + (rb + i) * 4;
  const float q0 = qp[0], q1 = qp[1], q2 = qp[2], q3 = qp[3];
  float m = -3.4e38f, l = 0.f;
  float acc[16];
#pragma unroll
  for (int d = 0; d < 16; ++d) acc[d] = 0.f;

  for (int j0 = 0; j0 < i; j0 += 32) {
    int j = j0 + lane;
    if (j < i) {                                      // strict causal: j < i
      const float* kp = k + (rb + j) * 4;
      float dot = (q0 * kp[0] + q1 * kp[1] + q2 * kp[2] + q3 * kp[3]) * 0.25f;
      if (dot > m) {
        float s = __expf(m - dot);
        l *= s;
#pragma unroll
        for (int d = 0; d < 16; ++d) acc[d] *= s;
        m = dot;
      }
      float pexp = __expf(dot - m);
      l += pexp;
      const float* vp = v + (rb + j) * 16;
#pragma unroll
      for (int d = 0; d < 16; ++d) acc[d] += pexp * vp[d];
    }
  }
  float mg = m;
#pragma unroll
  for (int off = 16; off > 0; off >>= 1) mg = fmaxf(mg, __shfl_xor(mg, off, 32));
  float s = (l > 0.f) ? __expf(m - mg) : 0.f;
  l *= s;
#pragma unroll
  for (int d = 0; d < 16; ++d) acc[d] *= s;
#pragma unroll
  for (int off = 16; off > 0; off >>= 1) {
    l += __shfl_xor(l, off, 32);
#pragma unroll
    for (int d = 0; d < 16; ++d) acc[d] += __shfl_xor(acc[d], off, 32);
  }
  if (lane == 0) {
    float inv = 1.f / (l + 1e-10f);
#pragma unroll
    for (int d = 0; d < 16; ++d) atten[(rb + i) * 16 + d] = acc[d] * inv;
  }
}

// ---------------- W3r[d,c] = sum_h W3[16h+d, c] (heads identical -> fold tile) ----------------
__global__ __launch_bounds__(256) void w3r_kernel(const float* __restrict__ W3,
                                                  float* __restrict__ W3r) {
  int i = blockIdx.x * blockDim.x + threadIdx.x;
  if (i >= 16 * 128) return;
  int d = i >> 7, c = i & 127;
  float s = 0.f;
#pragma unroll
  for (int h = 0; h < 8; ++h) s += W3[(size_t)(h * 16 + d) * 128 + c];
  W3r[i] = s;
}

// ---------------- out = x + c3 + atten @ W3r + b3 ----------------
__global__ __launch_bounds__(256) void final_kernel(const float* __restrict__ x,
    const float* __restrict__ c3, const float* __restrict__ atten,
    const float* __restrict__ W3r, const float* __restrict__ b3,
    float* __restrict__ out, int P) {
  int i = blockIdx.x * blockDim.x + threadIdx.x;
  if (i >= P * 128) return;
  int n = i >> 7, c = i & 127;
  float mh = b3[c];
  const float* ar = atten + (size_t)n * 16;
#pragma unroll
  for (int d = 0; d < 16; ++d) mh += ar[d] * W3r[d * 128 + c];
  out[i] = x[i] + c3[i] + mh;
}

extern "C" void kernel_launch(void* const* d_in, const int* in_sizes, int n_in,
                              void* d_out, int out_size, void* d_ws, size_t ws_size,
                              hipStream_t stream) {
  const float* x  = (const float*)d_in[0];
  const float* W1 = (const float*)d_in[1];
  const float* b1 = (const float*)d_in[2];
  const float* W2 = (const float*)d_in[3];
  const float* b2 = (const float*)d_in[4];
  const float* Wq = (const float*)d_in[5];
  const float* bq = (const float*)d_in[6];
  const float* Wk = (const float*)d_in[7];
  const float* bk = (const float*)d_in[8];
  const float* Wv = (const float*)d_in[9];
  const float* bv = (const float*)d_in[10];
  const float* W3 = (const float*)d_in[11];
  const float* b3 = (const float*)d_in[12];

  const int B = 4, H = 64, W = 64, C = 128;
  const int P = B * H * W;   // 16384
  const int N = H * W;       // 4096

  char* ws = (char*)d_ws;
  size_t off = 0;
  auto alloc = [&](size_t bytes) -> void* {
    void* p = (void*)(ws + off);
    off += (bytes + 255) & ~(size_t)255;
    return p;
  };
  unsigned short* ex  = (unsigned short*)alloc((size_t)P * C * 2);          // elu(x) bf16
  unsigned int*   W1p = (unsigned int*)  alloc((size_t)9 * C * C * 2);      // packed W1
  unsigned short* c1  = (unsigned short*)alloc((size_t)P * C * 2);          // conv1 out bf16
  unsigned int*   W2p = (unsigned int*)  alloc((size_t)9 * C * 2 * C * 2);  // packed W2
  float* c2  = (float*)alloc((size_t)P * 2 * C * 4);
  float* c3  = (float*)alloc((size_t)P * C * 4);
  float* qb  = (float*)alloc((size_t)P * 4 * 4);
  float* kb  = (float*)alloc((size_t)P * 4 * 4);
  float* vb  = (float*)alloc((size_t)P * 16 * 4);
  float* at  = (float*)alloc((size_t)P * 16 * 4);
  float* W3r = (float*)alloc((size_t)16 * 128 * 4);
  (void)ws_size; (void)in_sizes; (void)n_in; (void)out_size;

  const int nel = P * C;
  elu_to_bf16<<<(nel + 255) / 256, 256, 0, stream>>>(x, ex, nel);

  const int pw1 = 9 * (C / 16) * (C / 32) * 32 * 8;
  pack_weights<<<(pw1 + 255) / 256, 256, 0, stream>>>(W1, W1p, C, C);
  const int pw2 = 9 * (2 * C / 16) * (C / 32) * 32 * 8;
  pack_weights<<<(pw2 + 255) / 256, 256, 0, stream>>>(W2, W2p, C, 2 * C);

  conv3x3_wmma<128, 128, true ><<<(P / 128) * (128 / 64), 256, 0, stream>>>(
      ex, W1p, b1, (void*)c1, H, W);
  conv3x3_wmma<128, 256, false><<<(P / 128) * (256 / 64), 256, 0, stream>>>(
      c1, W2p, b2, (void*)c2, H, W);

  gate_kernel<<<(P * 128 + 255) / 256, 256, 0, stream>>>(c2, c3, P);
  qkv_kernel<<<(P * 24 + 255) / 256, 256, 0, stream>>>(c3, Wq, bq, Wk, bk, Wv, bv,
                                                       qb, kb, vb, P);
  attn_kernel<<<(P * 32) / 256, 256, 0, stream>>>(qb, kb, vb, at, N);
  w3r_kernel<<<(16 * 128 + 255) / 256, 256, 0, stream>>>(W3, W3r);
  final_kernel<<<(P * 128 + 255) / 256, 256, 0, stream>>>(x, c3, at, W3r, b3,
                                                          (float*)d_out, P);
}
